// MoELayer_20023137534915
// MI455X (gfx1250) — compile-verified, matrix operand order checked by reading, not run on hardware
//
#include <hip/hip_runtime.h>
#include <hip/hip_bf16.h>
#include <stdint.h>

// ---------------- problem constants ----------------
#define N_EMBD    1024
#define HIDDEN    2048
#define N_EXPERTS 8
#define NTOK      8192          // B*S
#define AUX_COEF  0.01f

// ---------------- FFN tiling ----------------
#define TM   64                 // slots per tile
#define HCW  128                // hidden chunk width
#define NHC  (HIDDEN / HCW)     // 16
#define KS1  128                // K slab for GEMM1 staging
#define KW   32                 // WMMA K
#define FFN_THREADS 512
#define MAX_TILES (NTOK * 2 / TM + N_EXPERTS)   // 264
#define MAX_SLOTS (MAX_TILES * TM)              // 16896

// LDS strides in fp16 elements (padded against bank conflicts)
#define XS   (N_EMBD + 8)       // 1032
#define W1S  (KS1 + 8)          // 136  (row = 272B, matches TDM pad 64DW+4DW)
#define HCS  (HCW + 8)          // 136
#define W2S  (KW + 4)           // 36   (row = 72B, matches TDM pad 16DW+2DW)

#define STAGE_BYTES (N_EMBD * W2S * 2)           // 73728: max(W1 slab 34816, W2 slab 73728)
#define LDS_X    0
#define LDS_B0   (LDS_X + TM * XS * 2)           // 132096
#define LDS_B1   (LDS_B0 + STAGE_BYTES)          // 205824
#define LDS_HC   (LDS_B1 + STAGE_BYTES)          // 279552
#define LDS_TOK  (LDS_HC + TM * HCS * 2)         // 296960
#define LDS_PRB  (LDS_TOK + 256)                 // 297216
#define SMEM_FFN (LDS_PRB + 256)                 // 297472 bytes  (< 320KB/WGP)

// ---------------- workspace layout (bytes) ----------------
#define WS_TOPK_IDX  0u                      // int[NTOK*2]
#define WS_TOPK_P    65536u                  // float[NTOK*2]
#define WS_SELCNT    131072u                 // int[8]
#define WS_FILL      131104u                 // int[8]
#define WS_OFFS      131136u                 // int[9]
#define WS_NTILES    131172u                 // int[1]
#define WS_TILE_E    131176u                 // int[264]
#define WS_TILE_S    132232u                 // int[264]
#define WS_CPART     133288u                 // int[1024*8]
#define WS_PPART     166056u                 // float[1024*8]
#define WS_TOKLIST   198912u                 // int[MAX_SLOTS]
#define WS_PROBLIST  266496u                 // float[MAX_SLOTS]
#define WS_W1T       334080u                 // fp16[8*2048*1024]
#define WS_W2T       33888512u               // fp16[8*1024*2048]
#define WS_TOTAL     67442944u

// ---------------- types ----------------
typedef __attribute__((ext_vector_type(16))) _Float16 v16h;
typedef __attribute__((ext_vector_type(8)))  float    v8f;
typedef __attribute__((ext_vector_type(4)))  unsigned int v4u;
typedef __attribute__((ext_vector_type(8)))  int      v8i;
typedef __attribute__((ext_vector_type(4)))  int      v4i;

union FragU { unsigned int u[8]; v16h v; };

#if defined(__HIP_DEVICE_COMPILE__) && defined(__gfx1250__) && __has_builtin(__builtin_amdgcn_tensor_load_to_lds)
#define HAS_TDM 1
#else
#define HAS_TDM 0
#endif

// ---------------- helpers ----------------
__device__ __forceinline__ unsigned short f2h1(float a) {
    union { _Float16 h; unsigned short u; } c; c.h = (_Float16)a; return c.u;
}
__device__ __forceinline__ unsigned f2h2(float a, float b) {
    return (unsigned)f2h1(a) | ((unsigned)f2h1(b) << 16);
}

// Load a 16x32 fp16 WMMA operand fragment from LDS.
// Tile origin at p0; per-lane: row/col = lane&15, K-halves selected by lane>>4
// (K base = sel*8, second K group at +16), each VGPR holds a contiguous K pair.
__device__ __forceinline__ v16h lds_frag(const char* p0, int strideB, int lane) {
    int lo = lane & 15, sel = lane >> 4;
    const char* p = p0 + lo * strideB + sel * 16;
    FragU f;
#pragma unroll
    for (int i = 0; i < 4; ++i) f.u[i]     = *(const unsigned int*)(p + 4 * i);
#pragma unroll
    for (int i = 0; i < 4; ++i) f.u[4 + i] = *(const unsigned int*)(p + 32 + 4 * i);
    return f.v;
}

#if HAS_TDM
// 2D tile (fp16) global -> LDS via Tensor Data Mover. Descriptor per ISA 08_async_tensor §8.
// Toolchain uses the 6-arg builtin: (g0 v4u, g1 v8i, g2 v4i, g3 v4i, g4 v8i, cpol i32).
__device__ __forceinline__ void tdm_load2d(unsigned ldsOff, const void* g,
                                           unsigned tileW, unsigned tileH,
                                           unsigned strideElems,
                                           unsigned tensorW, unsigned tensorH,
                                           unsigned padIntCode, unsigned padAmtCode) {
    unsigned long long ga = (unsigned long long)(uintptr_t)g;
    v4u g0;
    g0[0] = 1u;                                              // count=1, user desc, no gather
    g0[1] = ldsOff;                                          // lds_addr (bytes)
    g0[2] = (unsigned)(ga & 0xffffffffu);                    // global_addr[31:0]
    g0[3] = (unsigned)((ga >> 32) & 0x01ffffffu) | (2u << 30); // global_addr[56:32], type=2
    v8i g1;
    g1[0] = (int)((1u << 16) | (1u << 20) | (padIntCode << 22) | (padAmtCode << 25)); // 2B data, pad_en
    g1[1] = (int)((tensorW & 0xffffu) << 16);
    g1[2] = (int)((tensorW >> 16) | ((tensorH & 0xffffu) << 16));
    g1[3] = (int)((tensorH >> 16) | (tileW << 16));
    g1[4] = (int)tileH;                                      // tile_dim1 (tile_dim2=0)
    g1[5] = (int)strideElems;                                // tensor_dim0_stride
    g1[6] = 0; g1[7] = 0;
    v4i z4 = {0, 0, 0, 0};
    v8i z8 = {0, 0, 0, 0, 0, 0, 0, 0};
    __builtin_amdgcn_tensor_load_to_lds(g0, g1, z4, z4, z8, 0);
}

// Pipeline epoch ep of one hc iteration: ep 0..7 = W1 slabs, ep 8..11 = W2 slabs.
__device__ __forceinline__ void issue_stage(char* buf, int ep, int hc,
                                            const _Float16* w1e, const _Float16* w2e) {
    if (ep < 8) {
        tdm_load2d((unsigned)(uintptr_t)buf,
                   w1e + (size_t)(hc * HCW) * N_EMBD + ep * KS1,
                   KS1, HCW, N_EMBD, N_EMBD, HIDDEN,
                   /*pad: 64 DW interval*/5, /*4 DW amount*/3);
    } else {
        tdm_load2d((unsigned)(uintptr_t)buf,
                   w2e + hc * HCW + (ep - 8) * KW,
                   KW, N_EMBD, HIDDEN, HIDDEN, N_EMBD,
                   /*pad: 16 DW interval*/3, /*2 DW amount*/1);
    }
}
#else
__device__ __forceinline__ void coop_copy(char* dst, int dstStrideB, const _Float16* src,
                                          long srcStride, int rows, int cols, int tid, int nthr) {
    int chunks = cols >> 2;           // uint2 = 4 fp16
    int total = rows * chunks;
    for (int i = tid; i < total; i += nthr) {
        int r = i / chunks, c = i % chunks;
        *(uint2*)(dst + r * dstStrideB + c * 8) =
            *(const uint2*)(src + (size_t)r * srcStride + c * 4);
    }
}
#endif

// ---------------- kernel 0: init ----------------
__global__ __launch_bounds__(256) void moe_init_kernel(float* __restrict__ out,
                                                       int* __restrict__ sel_counts,
                                                       int* __restrict__ fill,
                                                       int* __restrict__ token_list,
                                                       float* __restrict__ prob_list) {
    size_t i = (size_t)blockIdx.x * blockDim.x + threadIdx.x;
    size_t stride = (size_t)gridDim.x * blockDim.x;
    for (size_t j = i; j < (size_t)NTOK * N_EMBD; j += stride) out[j] = 0.f;
    for (size_t j = i; j < MAX_SLOTS; j += stride) { token_list[j] = -1; prob_list[j] = 0.f; }
    if (i < N_EXPERTS) { sel_counts[i] = 0; fill[i] = 0; }
}

// ---------------- kernel 1: weight transpose + fp32->fp16 ----------------
// per expert z: src is R x C row-major fp32; dst is C x R fp16
__global__ __launch_bounds__(256) void transpose_f16_kernel(const float* __restrict__ src,
                                                            _Float16* __restrict__ dst,
                                                            int R, int C) {
    __shared__ float t[32][33];
    int e = blockIdx.z;
    int c0 = blockIdx.x * 32, r0 = blockIdx.y * 32;
    const float* s = src + (size_t)e * R * C;
    _Float16* d = dst + (size_t)e * R * C;
    int tx = threadIdx.x, ty = threadIdx.y;   // 32 x 8
#pragma unroll
    for (int i = 0; i < 4; ++i) {
        int r = r0 + ty + i * 8;
        t[ty + i * 8][tx] = s[(size_t)r * C + c0 + tx];
    }
    __syncthreads();
#pragma unroll
    for (int i = 0; i < 4; ++i) {
        int c = c0 + ty + i * 8;
        d[(size_t)c * R + r0 + tx] = (_Float16)t[tx][ty + i * 8];
    }
}

// ---------------- kernel 2: router (one wave32 per token) ----------------
__global__ __launch_bounds__(256) void router_kernel(const float* __restrict__ x,
                                                     const float* __restrict__ gate_w,
                                                     int* __restrict__ topk_idx,
                                                     float* __restrict__ topk_p,
                                                     int* __restrict__ sel_counts,
                                                     int* __restrict__ c_part,
                                                     float* __restrict__ p_part) {
    __shared__ float g[N_EXPERTS * N_EMBD];
    __shared__ float pP[8][N_EXPERTS];
    __shared__ int   pC[8][N_EXPERTS];
    int tid = threadIdx.x;
    for (int i = tid; i < N_EXPERTS * N_EMBD; i += 256) g[i] = gate_w[i];
    if (tid < 64) { pP[tid >> 3][tid & 7] = 0.f; pC[tid >> 3][tid & 7] = 0; }
    __syncthreads();
    int wave = tid >> 5, lane = tid & 31;
    int token = blockIdx.x * 8 + wave;
    float acc[N_EXPERTS];
#pragma unroll
    for (int e = 0; e < N_EXPERTS; ++e) acc[e] = 0.f;
    const float* xr = x + (size_t)token * N_EMBD;
    for (int d = lane; d < N_EMBD; d += 32) {
        float xv = xr[d];
#pragma unroll
        for (int e = 0; e < N_EXPERTS; ++e) acc[e] += xv * g[e * N_EMBD + d];
    }
#pragma unroll
    for (int e = 0; e < N_EXPERTS; ++e) {
        float v = acc[e];
        for (int off = 16; off > 0; off >>= 1) v += __shfl_xor(v, off, 32);
        acc[e] = v;
    }
    if (lane == 0) {
        float m = acc[0];
#pragma unroll
        for (int e = 1; e < N_EXPERTS; ++e) m = fmaxf(m, acc[e]);
        float p[N_EXPERTS], s = 0.f;
#pragma unroll
        for (int e = 0; e < N_EXPERTS; ++e) { p[e] = expf(acc[e] - m); s += p[e]; }
        float inv = 1.f / s;
#pragma unroll
        for (int e = 0; e < N_EXPERTS; ++e) p[e] *= inv;
        int i0 = 0;
#pragma unroll
        for (int e = 1; e < N_EXPERTS; ++e) if (p[e] > p[i0]) i0 = e;   // first max (jax tie rule)
        int i1 = (i0 == 0) ? 1 : 0;
#pragma unroll
        for (int e = 0; e < N_EXPERTS; ++e) if (e != i1 && e != i0 && p[e] > p[i1]) i1 = e;
        float ps = 1.f / (p[i0] + p[i1]);
        topk_idx[token * 2] = i0; topk_idx[token * 2 + 1] = i1;
        topk_p[token * 2] = p[i0] * ps; topk_p[token * 2 + 1] = p[i1] * ps;
        atomicAdd(&sel_counts[i0], 1);
        atomicAdd(&sel_counts[i1], 1);
#pragma unroll
        for (int e = 0; e < N_EXPERTS; ++e) pP[wave][e] = p[e];
        pC[wave][i0] = 1;
    }
    __syncthreads();
    if (tid < N_EXPERTS) {     // deterministic fixed-order block reduction
        float s = 0.f; int c = 0;
#pragma unroll
        for (int w = 0; w < 8; ++w) { s += pP[w][tid]; c += pC[w][tid]; }
        p_part[blockIdx.x * N_EXPERTS + tid] = s;
        c_part[blockIdx.x * N_EXPERTS + tid] = c;
    }
}

// ---------------- kernel 3: scan (offsets, tile table, aux loss) ----------------
__global__ void scan_kernel(const int* __restrict__ sel_counts,
                            const int* __restrict__ c_part,
                            const float* __restrict__ p_part,
                            int* __restrict__ offs, int* __restrict__ tile_e,
                            int* __restrict__ tile_s, int* __restrict__ n_tiles,
                            float* __restrict__ aux_out) {
    __shared__ float Ptot[N_EXPERTS];
    __shared__ int   Ctot[N_EXPERTS];
    int tid = threadIdx.x;
    if (tid < N_EXPERTS) {
        float s = 0.f; int c = 0;
        for (int b = 0; b < NTOK / 8; ++b) { s += p_part[b * N_EXPERTS + tid]; c += c_part[b * N_EXPERTS + tid]; }
        Ptot[tid] = s; Ctot[tid] = c;
    }
    __syncthreads();
    if (tid == 0) {
        int o = 0, t = 0;
        for (int e = 0; e < N_EXPERTS; ++e) {
            offs[e] = o;
            int nt = (sel_counts[e] + TM - 1) / TM;
            for (int i = 0; i < nt; ++i) { tile_e[t] = e; tile_s[t] = o + i * TM; ++t; }
            o += nt * TM;
        }
        offs[N_EXPERTS] = o;
        *n_tiles = t;
        const float invN = 1.f / (float)NTOK;
        float a = 0.f;
        for (int e = 0; e < N_EXPERTS; ++e) a += ((float)Ctot[e] * invN) * (Ptot[e] * invN);
        *aux_out = AUX_COEF * (float)N_EXPERTS * a;
    }
}

// ---------------- kernel 4: scatter tokens into expert-grouped slots ----------------
__global__ __launch_bounds__(256) void scatter_kernel(const int* __restrict__ topk_idx,
                                                      const float* __restrict__ topk_p,
                                                      const int* __restrict__ offs,
                                                      int* __restrict__ fill,
                                                      int* __restrict__ token_list,
                                                      float* __restrict__ prob_list) {
    int n = blockIdx.x * 256 + threadIdx.x;
    if (n >= NTOK) return;
#pragma unroll
    for (int k = 0; k < 2; ++k) {
        int e = topk_idx[n * 2 + k];
        float p = topk_p[n * 2 + k];
        int pos = atomicAdd(&fill[e], 1);
        int slot = offs[e] + pos;
        token_list[slot] = n;
        prob_list[slot] = p;
    }
}

// ---------------- kernel 5: fused expert FFN (GEMM1 -> ReLU -> GEMM2 -> combine) --------
// Double-buffered TDM pipeline: epoch e consumes buf[e&1] while epoch e+1's DMA streams
// into buf[(e+1)&1]; s_wait_tensorcnt(1) targets a DMA issued one full epoch earlier.
__global__ __launch_bounds__(FFN_THREADS) void ffn_kernel(const float* __restrict__ x,
                                                          const _Float16* __restrict__ w1t, // [E][H][D]
                                                          const _Float16* __restrict__ w2t, // [E][D][H]
                                                          const int* __restrict__ token_list,
                                                          const float* __restrict__ prob_list,
                                                          const int* __restrict__ tile_e,
                                                          const int* __restrict__ tile_s,
                                                          const int* __restrict__ n_tiles,
                                                          float* __restrict__ out) {
    extern __shared__ char smem[];
    char*  Xl   = smem + LDS_X;
    char*  B0   = smem + LDS_B0;
    char*  B1   = smem + LDS_B1;
    char*  Hcb  = smem + LDS_HC;
    int*   tokL = (int*)(smem + LDS_TOK);
    float* prL  = (float*)(smem + LDS_PRB);

    int b = blockIdx.x;
    if (b >= *n_tiles) return;
    int e = tile_e[b];
    int s0 = tile_s[b];
    int tid = threadIdx.x;
    int wave = tid >> 5, lane = tid & 31;
    int sel = lane >> 4, lo = lane & 15;

    const _Float16* w1e = w1t + (size_t)e * HIDDEN * N_EMBD;
    const _Float16* w2e = w2t + (size_t)e * N_EMBD * HIDDEN;

#if HAS_TDM
    if (tid < 32) issue_stage(B0, 0, 0, w1e, w2e);   // pipeline preamble: first W1 slab
#endif

    if (tid < TM) { tokL[tid] = token_list[s0 + tid]; prL[tid] = prob_list[s0 + tid]; }
    __syncthreads();

    // ---- gather X tile (64 tokens x 1024) fp32 -> fp16 in LDS ----
    {
        int r = tid >> 3, c8 = tid & 7;
        int token = tokL[r];
        const float* xr = x + (size_t)(token < 0 ? 0 : token) * N_EMBD;
        char* dstRow = Xl + r * (XS * 2);
#pragma unroll 4
        for (int j = 0; j < 32; ++j) {
            int col = c8 * 128 + j * 4;
            float4 v = (token >= 0) ? *(const float4*)(xr + col) : make_float4(0.f, 0.f, 0.f, 0.f);
            uint2 pk;
            pk.x = f2h2(v.x, v.y);
            pk.y = f2h2(v.z, v.w);
            *(uint2*)(dstRow + col * 2) = pk;
        }
    }

    int mt  = wave & 3;    // phase-1 row tile (16 slots)
    int nt2 = wave >> 2;   // phase-1 col pair (32 of the 128 hidden cols)
    int gpar = 0;          // parity of current pipeline epoch

    v8f zero = {0.f, 0.f, 0.f, 0.f, 0.f, 0.f, 0.f, 0.f};
    v8f acc2[4][4];
#pragma unroll
    for (int m = 0; m < 4; ++m)
#pragma unroll
        for (int n = 0; n < 4; ++n) acc2[m][n] = zero;

    for (int hc = 0; hc < NHC; ++hc) {
        // ======== phase 1: Hc = relu(X @ W1[:, hc*128 : +128]) ========
        v8f a0 = zero, a1 = zero;
        for (int j = 0; j < 8; ++j) {
            char* cur = gpar ? B1 : B0;
            char* nxt = gpar ? B0 : B1;
            __syncthreads();                 // nxt's previous consumers are done
#if HAS_TDM
            if (tid < 32) {
                issue_stage(nxt, j + 1, hc, w1e, w2e);   // j==7 prefetches W2 slab 0
                __builtin_amdgcn_s_wait_tensorcnt(1);    // epoch j's DMA complete
            }
#else
            coop_copy(cur, W1S * 2, w1e + (size_t)(hc * HCW) * N_EMBD + j * KS1,
                      N_EMBD, HCW, KS1, tid, FFN_THREADS);
#endif
            __syncthreads();                 // cur visible to all waves
            int ks = j * KS1;
#pragma unroll
            for (int kk = 0; kk < KS1; kk += KW) {
                v16h af = lds_frag(Xl + (mt * 16) * (XS * 2) + (ks + kk) * 2, XS * 2, lane);
                v16h b0 = lds_frag(cur + (nt2 * 32) * (W1S * 2) + kk * 2, W1S * 2, lane);
                v16h b1 = lds_frag(cur + (nt2 * 32 + 16) * (W1S * 2) + kk * 2, W1S * 2, lane);
                a0 = __builtin_amdgcn_wmma_f32_16x16x32_f16(false, af, false, b0, (short)0, a0, false, false);
                a1 = __builtin_amdgcn_wmma_f32_16x16x32_f16(false, af, false, b1, (short)0, a1, false, false);
            }
            gpar ^= 1;
        }
        // relu + fp16 into Hc LDS (old-Hc readers finished many barriers ago)
#pragma unroll
        for (int v = 0; v < 8; ++v) {
            int row = mt * 16 + sel * 8 + v;
            char* hp = Hcb + row * (HCS * 2);
            float v0 = a0[v] > 0.f ? a0[v] : 0.f;
            float v1 = a1[v] > 0.f ? a1[v] : 0.f;
            *(unsigned short*)(hp + (nt2 * 32 + lo) * 2)      = f2h1(v0);
            *(unsigned short*)(hp + (nt2 * 32 + 16 + lo) * 2) = f2h1(v1);
        }
        // ======== phase 2: acc2 += Hc @ W2[hc*128 : +128, :] ========
        for (int s = 0; s < 4; ++s) {
            char* cur = gpar ? B1 : B0;
            char* nxt = gpar ? B0 : B1;
            __syncthreads();                 // Hc writes visible; nxt's consumers done
#if HAS_TDM
            if (tid < 32) {
                // s_wait_tensorcnt needs a literal immediate -> constant in each branch
                if (s < 3) {
                    issue_stage(nxt, 9 + s, hc, w1e, w2e);      // next W2 slab
                    __builtin_amdgcn_s_wait_tensorcnt(1);
                } else if (hc < NHC - 1) {
                    issue_stage(nxt, 0, hc + 1, w1e, w2e);      // next hc W1 slab 0
                    __builtin_amdgcn_s_wait_tensorcnt(1);
                } else {
                    __builtin_amdgcn_s_wait_tensorcnt(0);       // drain final epoch
                }
            }
#else
            coop_copy(cur, W2S * 2, w2e + hc * HCW + s * KW, HIDDEN, N_EMBD, KW, tid, FFN_THREADS);
#endif
            __syncthreads();                 // cur visible to all waves
            int ks2 = s * KW;
            v16h am[4];
#pragma unroll
            for (int m = 0; m < 4; ++m)
                am[m] = lds_frag(Hcb + (m * 16) * (HCS * 2) + ks2 * 2, HCS * 2, lane);
#pragma unroll
            for (int n = 0; n < 4; ++n) {
                v16h bf = lds_frag(cur + (wave * 64 + n * 16) * (W2S * 2), W2S * 2, lane);
#pragma unroll
                for (int m = 0; m < 4; ++m)
                    acc2[m][n] = __builtin_amdgcn_wmma_f32_16x16x32_f16(false, am[m], false, bf,
                                                                        (short)0, acc2[m][n], false, false);
            }
            gpar ^= 1;
        }
    }

    // ---- combine: out[token] += p * y  (exactly 2 addends/elem -> deterministic) ----
#pragma unroll
    for (int m = 0; m < 4; ++m) {
#pragma unroll
        for (int v = 0; v < 8; ++v) {
            int r = m * 16 + sel * 8 + v;
            int token = tokL[r];
            if (token < 0) continue;
            float p = prL[r];
            float* orow = out + (size_t)token * N_EMBD + wave * 64;
#pragma unroll
            for (int n = 0; n < 4; ++n)
                unsafeAtomicAdd(orow + n * 16 + lo, acc2[m][n][v] * p);
        }
    }
}

// ---------------- launcher ----------------
extern "C" void kernel_launch(void* const* d_in, const int* in_sizes, int n_in,
                              void* d_out, int out_size, void* d_ws, size_t ws_size,
                              hipStream_t stream) {
    (void)in_sizes; (void)n_in; (void)out_size;
    if (ws_size < WS_TOTAL) return;

    const float* x      = (const float*)d_in[0];
    const float* gate_w = (const float*)d_in[1];
    const float* w1     = (const float*)d_in[2];
    const float* w2     = (const float*)d_in[3];
    float* out = (float*)d_out;
    char*  ws  = (char*)d_ws;

    int*      topk_idx   = (int*)(ws + WS_TOPK_IDX);
    float*    topk_p     = (float*)(ws + WS_TOPK_P);
    int*      sel_counts = (int*)(ws + WS_SELCNT);
    int*      fill       = (int*)(ws + WS_FILL);
    int*      offs       = (int*)(ws + WS_OFFS);
    int*      n_tiles    = (int*)(ws + WS_NTILES);
    int*      tile_e     = (int*)(ws + WS_TILE_E);
    int*      tile_s     = (int*)(ws + WS_TILE_S);
    int*      c_part     = (int*)(ws + WS_CPART);
    float*    p_part     = (float*)(ws + WS_PPART);
    int*      token_list = (int*)(ws + WS_TOKLIST);
    float*    prob_list  = (float*)(ws + WS_PROBLIST);
    _Float16* w1t        = (_Float16*)(ws + WS_W1T);
    _Float16* w2t        = (_Float16*)(ws + WS_W2T);

    (void)hipFuncSetAttribute(reinterpret_cast<const void*>(ffn_kernel),
                              hipFuncAttributeMaxDynamicSharedMemorySize, SMEM_FFN);

    moe_init_kernel<<<2048, 256, 0, stream>>>(out, sel_counts, fill, token_list, prob_list);
    // w1: (E, D=1024, H=2048) -> (E, H, D) fp16
    transpose_f16_kernel<<<dim3(HIDDEN / 32, N_EMBD / 32, N_EXPERTS), dim3(32, 8), 0, stream>>>(
        w1, w1t, N_EMBD, HIDDEN);
    // w2: (E, H=2048, D=1024) -> (E, D, H) fp16
    transpose_f16_kernel<<<dim3(N_EMBD / 32, HIDDEN / 32, N_EXPERTS), dim3(32, 8), 0, stream>>>(
        w2, w2t, HIDDEN, N_EMBD);
    router_kernel<<<NTOK / 8, 256, 0, stream>>>(x, gate_w, topk_idx, topk_p,
                                                sel_counts, c_part, p_part);
    scan_kernel<<<1, 256, 0, stream>>>(sel_counts, c_part, p_part, offs, tile_e, tile_s,
                                       n_tiles, out + (size_t)NTOK * N_EMBD);
    scatter_kernel<<<NTOK / 256, 256, 0, stream>>>(topk_idx, topk_p, offs, fill,
                                                   token_list, prob_list);
    ffn_kernel<<<MAX_TILES, FFN_THREADS, SMEM_FFN, stream>>>(x, w1t, w2t, token_list, prob_list,
                                                             tile_e, tile_s, n_tiles, out);
}
